// S4Layer_35510789603610
// MI455X (gfx1250) — compile-verified, matrix operand order checked by reading, not run on hardware
//
#include <hip/hip_runtime.h>
#include <hip/hip_bf16.h>

#define B_SZ 4
#define L_SZ 4096
#define H_SZ 1024
#define N_SZ 64
#define P_SZ 64                 // chunk length
#define C_SZ (L_SZ / P_SZ)      // 64 chunks
#define NCOL (B_SZ * C_SZ)      // 256 matmul columns per h
#define LN_EPS 1e-5f

typedef __attribute__((ext_vector_type(16))) _Float16 v16h;
typedef __attribute__((ext_vector_type(8)))  _Float16 v8h;
typedef __attribute__((ext_vector_type(8)))  float    v8f;
typedef __attribute__((ext_vector_type(4)))  float    v4f;

// ---------------------------------------------------------------------------
// Kernel 1: per-h parameter precompute (discretization, Toeplitz kernel,
// ingest matrix Win, emission matrix E, per-chunk decay a^P).
// One block (64 threads) per h.
// ---------------------------------------------------------------------------
__global__ void precompute_kernel(const float* __restrict__ log_A_real,
                                  const float* __restrict__ A_imag,
                                  const float* __restrict__ C_re,
                                  const float* __restrict__ C_im,
                                  const float* __restrict__ log_dt,
                                  _Float16* __restrict__ Win,
                                  _Float16* __restrict__ Emat,
                                  _Float16* __restrict__ Tmat,
                                  float* __restrict__ decayP) {
  __shared__ float s_dtAr[N_SZ], s_dtAi[N_SZ], s_Ctr[N_SZ], s_Cti[N_SZ], s_K[P_SZ];
  const int h = blockIdx.x;
  const int t = threadIdx.x;                 // 0..63, doubles as state index n
  const float dt = __expf(log_dt[h]);

  // thread t owns state n = t for the setup phase
  const float Ar = -__expf(log_A_real[h * N_SZ + t]);
  const float Ai = A_imag[h * N_SZ + t];
  const float dAr = dt * Ar, dAi = dt * Ai;
  const float e0 = __expf(dAr);
  const float are = e0 * __cosf(dAi), aim = e0 * __sinf(dAi);
  // Ct = C * (exp(dtA)-1)/A
  const float xr = are - 1.0f, xi = aim;
  const float inv = 1.0f / (Ar * Ar + Ai * Ai);
  const float qr = (xr * Ar + xi * Ai) * inv;
  const float qi = (xi * Ar - xr * Ai) * inv;
  const float cr = C_re[h * N_SZ + t], ci = C_im[h * N_SZ + t];
  const float Ctr = cr * qr - ci * qi;
  const float Cti = cr * qi + ci * qr;
  s_dtAr[t] = dAr; s_dtAi[t] = dAi; s_Ctr[t] = Ctr; s_Cti[t] = Cti;

  // per-chunk decay a^P
  const float eP = __expf(dAr * (float)P_SZ);
  decayP[(h * N_SZ + t) * 2 + 0] = eP * __cosf(dAi * (float)P_SZ);
  decayP[(h * N_SZ + t) * 2 + 1] = eP * __sinf(dAi * (float)P_SZ);
  __syncthreads();

  // K[d] = 2 * sum_n Re(Ct_n * a_n^d) ; thread t computes d = t
  {
    float acc = 0.0f;
    for (int n = 0; n < N_SZ; ++n) {
      const float ed = __expf(s_dtAr[n] * (float)t);
      const float cc = __cosf(s_dtAi[n] * (float)t);
      const float ss = __sinf(s_dtAi[n] * (float)t);
      acc += 2.0f * (s_Ctr[n] * ed * cc - s_Cti[n] * ed * ss);
    }
    s_K[t] = acc;
  }

  // Win[h][n][t]     = Re(a_n^(P-1-t))   (row-major, ld = P)
  // Win[h][N+n][t]   = Im(a_n^(P-1-t))
  {
    const float pw = (float)(P_SZ - 1 - t);
    _Float16* wp = Win + (size_t)h * (2 * N_SZ * P_SZ);
    for (int n = 0; n < N_SZ; ++n) {
      const float ed = __expf(s_dtAr[n] * pw);
      wp[n * P_SZ + t]          = (_Float16)(ed * __cosf(s_dtAi[n] * pw));
      wp[(N_SZ + n) * P_SZ + t] = (_Float16)(ed * __sinf(s_dtAi[n] * pw));
    }
  }

  // E[h][j][n]   =  2*Re(Ct_n a_n^(j+1))  (row-major, ld = 2N), thread j = t
  // E[h][j][N+n] = -2*Im(Ct_n a_n^(j+1))
  {
    const float pw = (float)(t + 1);
    _Float16* ep = Emat + (size_t)h * (P_SZ * 2 * N_SZ) + (size_t)t * (2 * N_SZ);
    for (int n = 0; n < N_SZ; ++n) {
      const float ed = __expf(s_dtAr[n] * pw);
      const float cc = __cosf(s_dtAi[n] * pw), ss = __sinf(s_dtAi[n] * pw);
      const float pr = s_Ctr[n] * ed * cc - s_Cti[n] * ed * ss;
      const float pi = s_Ctr[n] * ed * ss + s_Cti[n] * ed * cc;
      ep[n]        = (_Float16)( 2.0f * pr);
      ep[N_SZ + n] = (_Float16)(-2.0f * pi);
    }
  }
  __syncthreads();

  // Lower-triangular Toeplitz T[h][j][u] = (u<=j) ? K[j-u] : 0 ; thread j = t
  {
    _Float16* tp = Tmat + (size_t)h * (P_SZ * P_SZ) + (size_t)t * P_SZ;
    for (int u = 0; u < P_SZ; ++u)
      tp[u] = (u <= t) ? (_Float16)s_K[t - u] : (_Float16)0.0f;
  }
}

// ---------------------------------------------------------------------------
// Kernel 2: pack x [B,L,H] f32 -> Xc[h][col][t] f16 via LDS transpose.
// Block = 256 threads handles one (b, c) chunk x 64-h tile (64x64 elements x16).
// Coalesced f32 reads along h; contiguous f16 writes along t.
// ---------------------------------------------------------------------------
__global__ void pack_x_kernel(const float* __restrict__ x, _Float16* __restrict__ Xc) {
  __shared__ _Float16 s_tile[64 * 65];
  const int col = blockIdx.x;              // c*B + b
  const int hb  = blockIdx.y;              // h tile (0..15)
  const int b   = col & (B_SZ - 1);
  const int c   = col >> 2;
  const int tid = threadIdx.x;
#pragma unroll
  for (int i = 0; i < 16; ++i) {
    const int lin = i * 256 + tid;
    const int t  = lin >> 6;               // position within chunk
    const int hh = lin & 63;               // h within tile (fast -> coalesced)
    const int l  = c * P_SZ + t;
    s_tile[hh * 65 + t] =
        (_Float16)x[((size_t)b * L_SZ + l) * H_SZ + hb * 64 + hh];
  }
  __syncthreads();
#pragma unroll
  for (int i = 0; i < 16; ++i) {
    const int lin = i * 256 + tid;
    const int hh = lin >> 6;
    const int t  = lin & 63;               // t fast -> contiguous writes
    const int h  = hb * 64 + hh;
    Xc[(size_t)h * (NCOL * P_SZ) + (size_t)col * P_SZ + t] = s_tile[hh * 65 + t];
  }
}

// ---------------------------------------------------------------------------
// WMMA fragment loader. Row-major A (row per lane) and column-contiguous B
// (col per lane) share the identical per-lane K packing:
//   halves 0..7  -> K = k0 + (lane<16 ? 0 : 8) + i
//   halves 8..15 -> K = k0 + 16 + (lane<16 ? 0 : 8) + i
// All runs are 8 contiguous halves (16B aligned) -> b128 loads.
// ---------------------------------------------------------------------------
__device__ __forceinline__ v16h load_frag(const _Float16* base, int ld, int rc0,
                                          int k0, int lane) {
  const _Float16* p = base + (size_t)(rc0 + (lane & 15)) * ld + k0 + ((lane & 16) ? 8 : 0);
  v16h f;
#pragma unroll
  for (int i = 0; i < 8; ++i) { f[i] = p[i]; f[8 + i] = p[16 + i]; }
  return f;
}

// ---------------------------------------------------------------------------
// Kernel 3: ingest matmul. Per h: U[col][row] (col-contiguous, ld=2N) with
// U_logical[128 x 256] = Win[128 x 64] * X[64 x 256].
// grid (H, 16), block 256 (8 waves); one 16x16 tile per wave.
// Each lane's 8 accumulator rows are contiguous in U -> single b128 store.
// ---------------------------------------------------------------------------
__global__ void ingest_kernel(const _Float16* __restrict__ Win,
                              const _Float16* __restrict__ Xc,
                              _Float16* __restrict__ U) {
  const int h    = blockIdx.x;
  const int wave = threadIdx.x >> 5;
  const int lane = threadIdx.x & 31;
  const int tile = blockIdx.y * 8 + wave;      // 0..127
  const int m = tile >> 4, n = tile & 15;
  const _Float16* A  = Win + (size_t)h * (2 * N_SZ * P_SZ);
  const _Float16* Bm = Xc  + (size_t)h * (NCOL * P_SZ);
  v8f acc = {};
#pragma unroll
  for (int k0 = 0; k0 < P_SZ; k0 += 32) {
    __builtin_prefetch(Bm + (size_t)(n * 16) * P_SZ + k0 + 32, 0, 1);
    v16h a = load_frag(A,  P_SZ, m * 16, k0, lane);
    v16h b = load_frag(Bm, P_SZ, n * 16, k0, lane);
    acc = __builtin_amdgcn_wmma_f32_16x16x32_f16(false, a, false, b, (short)0, acc,
                                                 false, false);
  }
  _Float16* Up = U + (size_t)h * (2 * N_SZ * NCOL);
  const int colg  = n * 16 + (lane & 15);
  const int rbase = m * 16 + ((lane & 16) ? 8 : 0);
  v8h hv;
#pragma unroll
  for (int i = 0; i < 8; ++i) hv[i] = (_Float16)acc[i];
  *(v8h*)(Up + (size_t)colg * (2 * N_SZ) + rbase) = hv;   // 16B store
}

// ---------------------------------------------------------------------------
// Kernel 4: inter-chunk recurrence. One thread per (b,h,n) complex state:
// emit state-before-chunk into Sh[h][col][0..2N) (f16, col-contiguous),
// then s <- a^P * s + u. U reads are coalesced (consecutive n -> consecutive).
// ---------------------------------------------------------------------------
__global__ void scan_kernel(const _Float16* __restrict__ U,
                            const float* __restrict__ decayP,
                            _Float16* __restrict__ Sh) {
  const int idx = blockIdx.x * blockDim.x + threadIdx.x;   // B*H*N
  const int n   = idx & (N_SZ - 1);
  const int tmp = idx >> 6;
  const int b   = tmp & (B_SZ - 1);
  const int h   = tmp >> 2;
  const float dr = decayP[(h * N_SZ + n) * 2 + 0];
  const float di = decayP[(h * N_SZ + n) * 2 + 1];
  const _Float16* Up = U  + (size_t)h * (2 * N_SZ * NCOL);
  _Float16*       Sp = Sh + (size_t)h * (NCOL * 2 * N_SZ);
  float sr = 0.0f, si = 0.0f;
  for (int c = 0; c < C_SZ; ++c) {
    const int col = c * B_SZ + b;
    Sp[(size_t)col * (2 * N_SZ) + n]        = (_Float16)sr;
    Sp[(size_t)col * (2 * N_SZ) + N_SZ + n] = (_Float16)si;
    const float ur = (float)Up[(size_t)col * (2 * N_SZ) + n];
    const float ui = (float)Up[(size_t)col * (2 * N_SZ) + N_SZ + n];
    const float nsr = dr * sr - di * si + ur;
    const float nsi = dr * si + di * sr + ui;
    sr = nsr; si = nsi;
  }
}

// ---------------------------------------------------------------------------
// Kernel 5: emit matmul. Per h (Y stored [h][col][j], col-contiguous, ld=P):
//   Y_logical[64 x 256] = T[64 x 64] * X[64 x 256]   (intra-chunk causal conv)
//                       + E[64 x 128] * S[128 x 256] (inter-chunk emission)
// grid (H, 8), block 256 (8 waves); one 16x16 tile per wave.
// Lane stores 8 contiguous f32 -> two b128 stores.
// ---------------------------------------------------------------------------
__global__ void emit_kernel(const _Float16* __restrict__ Tmat,
                            const _Float16* __restrict__ Emat,
                            const _Float16* __restrict__ Xc,
                            const _Float16* __restrict__ Sh,
                            float* __restrict__ Y) {
  const int h    = blockIdx.x;
  const int wave = threadIdx.x >> 5;
  const int lane = threadIdx.x & 31;
  const int tile = blockIdx.y * 8 + wave;      // 0..63
  const int m = tile >> 4, n = tile & 15;      // m: 0..3
  const _Float16* At = Tmat + (size_t)h * (P_SZ * P_SZ);
  const _Float16* Ae = Emat + (size_t)h * (P_SZ * 2 * N_SZ);
  const _Float16* Bx = Xc   + (size_t)h * (NCOL * P_SZ);
  const _Float16* Bs = Sh   + (size_t)h * (NCOL * 2 * N_SZ);
  v8f acc = {};
#pragma unroll
  for (int k0 = 0; k0 < P_SZ; k0 += 32) {
    v16h a = load_frag(At, P_SZ, m * 16, k0, lane);
    v16h b = load_frag(Bx, P_SZ, n * 16, k0, lane);
    acc = __builtin_amdgcn_wmma_f32_16x16x32_f16(false, a, false, b, (short)0, acc,
                                                 false, false);
  }
#pragma unroll
  for (int k0 = 0; k0 < 2 * N_SZ; k0 += 32) {
    v16h a = load_frag(Ae, 2 * N_SZ, m * 16, k0, lane);
    v16h b = load_frag(Bs, 2 * N_SZ, n * 16, k0, lane);
    acc = __builtin_amdgcn_wmma_f32_16x16x32_f16(false, a, false, b, (short)0, acc,
                                                 false, false);
  }
  float* Yp = Y + (size_t)h * (P_SZ * NCOL);
  const int colg  = n * 16 + (lane & 15);
  const int rbase = m * 16 + ((lane & 16) ? 8 : 0);
  float* dst = Yp + (size_t)colg * P_SZ + rbase;
  v4f lo = {acc[0], acc[1], acc[2], acc[3]};
  v4f hi = {acc[4], acc[5], acc[6], acc[7]};
  *(v4f*)dst       = lo;                        // 16B store
  *(v4f*)(dst + 4) = hi;                        // 16B store
}

// ---------------------------------------------------------------------------
// Kernel 6: transpose Y[h][col][j] -> R[b,l,h] with fused residual:
//   r = x*(1+D) + y
// Block = 256 threads per (col, h-tile): coalesced f32 reads along j,
// coalesced writes (and x reads) along h.
// ---------------------------------------------------------------------------
__global__ void transpose_residual_kernel(const float* __restrict__ Y,
                                          const float* __restrict__ x,
                                          const float* __restrict__ D,
                                          float* __restrict__ R) {
  __shared__ float s_tile[64 * 65];
  const int col = blockIdx.x;              // c*B + b
  const int hb  = blockIdx.y;              // h tile (0..15)
  const int b   = col & (B_SZ - 1);
  const int c   = col >> 2;
  const int tid = threadIdx.x;
#pragma unroll
  for (int i = 0; i < 16; ++i) {
    const int lin = i * 256 + tid;
    const int hh = lin >> 6;
    const int j  = lin & 63;               // j fast -> coalesced reads
    s_tile[hh * 65 + j] =
        Y[((size_t)(hb * 64 + hh) * NCOL + col) * P_SZ + j];
  }
  __syncthreads();
#pragma unroll
  for (int i = 0; i < 16; ++i) {
    const int lin = i * 256 + tid;
    const int j  = lin >> 6;
    const int hh = lin & 63;               // h fast -> coalesced writes
    const int h  = hb * 64 + hh;
    const size_t bl = (size_t)b * L_SZ + c * P_SZ + j;
    const float xv = x[bl * H_SZ + h];
    R[bl * H_SZ + h] = s_tile[hh * 65 + j] + xv * (1.0f + D[h]);
  }
}

// ---------------------------------------------------------------------------
// Kernel 7: LayerNorm over H. One block per (b,l); fully coalesced.
// out = (r-mu)*rsqrt(var+eps)*w + b
// ---------------------------------------------------------------------------
__global__ void final_kernel(const float* __restrict__ R,
                             const float* __restrict__ ln_w,
                             const float* __restrict__ ln_b,
                             float* __restrict__ out) {
  __shared__ float s_sum[256], s_sq[256];
  const int bl  = blockIdx.x;              // b*L + l
  const int tid = threadIdx.x;
  float r[4];
  float lsum = 0.0f, lsq = 0.0f;
#pragma unroll
  for (int i = 0; i < 4; ++i) {
    const int h  = i * 256 + tid;
    const float rv = R[(size_t)bl * H_SZ + h];
    r[i] = rv; lsum += rv; lsq += rv * rv;
  }
  s_sum[tid] = lsum; s_sq[tid] = lsq;
  __syncthreads();
  for (int off = 128; off > 0; off >>= 1) {
    if (tid < off) { s_sum[tid] += s_sum[tid + off]; s_sq[tid] += s_sq[tid + off]; }
    __syncthreads();
  }
  const float mu  = s_sum[0] * (1.0f / (float)H_SZ);
  const float var = s_sq[0] * (1.0f / (float)H_SZ) - mu * mu;
  const float inv = rsqrtf(var + LN_EPS);
#pragma unroll
  for (int i = 0; i < 4; ++i) {
    const int h = i * 256 + tid;
    out[(size_t)bl * H_SZ + h] = (r[i] - mu) * inv * ln_w[h] + ln_b[h];
  }
}

// ---------------------------------------------------------------------------
// Host launcher
// ---------------------------------------------------------------------------
extern "C" void kernel_launch(void* const* d_in, const int* in_sizes, int n_in,
                              void* d_out, int out_size, void* d_ws, size_t ws_size,
                              hipStream_t stream) {
  const float* x          = (const float*)d_in[0];
  const float* log_A_real = (const float*)d_in[1];
  const float* A_imag     = (const float*)d_in[2];
  const float* C_re       = (const float*)d_in[3];
  const float* C_im       = (const float*)d_in[4];
  const float* log_dt     = (const float*)d_in[5];
  const float* Dterm      = (const float*)d_in[6];
  const float* ln_w       = (const float*)d_in[7];
  const float* ln_b       = (const float*)d_in[8];
  float* out = (float*)d_out;

  // workspace layout (bytes)
  char* ws = (char*)d_ws;
  const size_t WIN_B = (size_t)H_SZ * 2 * N_SZ * P_SZ * 2;   // 16 MB
  const size_t E_B   = (size_t)H_SZ * P_SZ * 2 * N_SZ * 2;   // 16 MB
  const size_t T_B   = (size_t)H_SZ * P_SZ * P_SZ * 2;       //  8 MB
  const size_t DEC_B = (size_t)H_SZ * N_SZ * 2 * 4;          // 0.5 MB
  const size_t XC_B  = (size_t)H_SZ * NCOL * P_SZ * 2;       // 32 MB
  const size_t U_B   = (size_t)H_SZ * 2 * N_SZ * NCOL * 2;   // 64 MB
  const size_t SH_B  = (size_t)H_SZ * NCOL * 2 * N_SZ * 2;   // 64 MB
  size_t off = 0;
  _Float16* Win    = (_Float16*)(ws + off); off += WIN_B;
  _Float16* Emat   = (_Float16*)(ws + off); off += E_B;
  _Float16* Tmat   = (_Float16*)(ws + off); off += T_B;
  float*    decayP = (float*)   (ws + off); off += DEC_B;
  _Float16* Xc     = (_Float16*)(ws + off); off += XC_B;
  _Float16* U      = (_Float16*)(ws + off); // U dead after scan -> Y aliases it
  float*    Yconv  = (float*)   (ws + off); off += U_B;
  _Float16* Sh     = (_Float16*)(ws + off); off += SH_B;
  // R (r = x*(1+D)+y, [B,L,H] f32, 67 MB) aliases Win..Xc (72.5 MB, dead
  // after emit_kernel).
  float* R = (float*)ws;
  (void)ws_size; (void)n_in; (void)in_sizes; (void)out_size;

  // 1) parameter precompute: 1024 blocks x 64 threads
  precompute_kernel<<<dim3(H_SZ), dim3(P_SZ), 0, stream>>>(
      log_A_real, A_imag, C_re, C_im, log_dt, Win, Emat, Tmat, decayP);

  // 2) pack x -> f16 column tiles (LDS transpose): grid (256 cols, 16 h-tiles)
  pack_x_kernel<<<dim3(NCOL, H_SZ / 64), dim3(256), 0, stream>>>(x, Xc);

  // 3) ingest matmul (WMMA): per h 128 tiles -> grid (H, 16) x 8 waves
  ingest_kernel<<<dim3(H_SZ, 16), dim3(256), 0, stream>>>(Win, Xc, U);

  // 4) inter-chunk scan: B*H*N threads
  scan_kernel<<<dim3((B_SZ * H_SZ * N_SZ) / 256), dim3(256), 0, stream>>>(
      U, decayP, Sh);

  // 5) emit matmul (WMMA, Y aliases U): per h 64 tiles -> grid (H, 8) x 8 waves
  emit_kernel<<<dim3(H_SZ, 8), dim3(256), 0, stream>>>(Tmat, Emat, Xc, Sh, Yconv);

  // 6) transpose back + fused residual (R aliases Win..Xc region)
  transpose_residual_kernel<<<dim3(NCOL, H_SZ / 64), dim3(256), 0, stream>>>(
      Yconv, x, Dterm, R);

  // 7) LayerNorm: one block per (b,l), fully coalesced
  final_kernel<<<dim3(B_SZ * L_SZ), dim3(256), 0, stream>>>(R, ln_w, ln_b, out);
}
